// DeepSeekNeuralAttention_55791625175280
// MI455X (gfx1250) — matrix operand order repaired and checked by reading, not compiled, not run on hardware
//
#include <hip/hip_runtime.h>

typedef __attribute__((ext_vector_type(16))) _Float16 v16h;
typedef __attribute__((ext_vector_type(8)))  _Float16 v8h;
typedef __attribute__((ext_vector_type(8)))  float    v8f;

#define HIDDEN 2048
#define NHEADS 16
#define HDIM   128
#define BATCH  2
#define SEQ    2048
#define NCP    32540
#define MTOT   (BATCH*SEQ)        // 4096 rows
#define WTOT   (HIDDEN*HIDDEN)    // 4194304 weight elements

// Workgroup barrier that drains only DScnt (LDS), NOT loadcnt: lets register
// prefetch global loads stay in flight across the barrier. All call sites are
// in workgroup-uniform control flow.
__device__ __forceinline__ void wg_barrier_lds() {
  asm volatile("s_wait_dscnt 0x0\n\t"
               "s_barrier_signal -1\n\t"
               "s_barrier_wait -1" ::: "memory");
}

// ---------- fragment loaders (ISA 7.12.2 layouts, wave32) ----------
// A-frag 16x32 f16: lane l holds row l%16; slot j -> k = 16*(j/8) + 8*(l/16) + (j%8)
__device__ __forceinline__ v16h load_a_frag(const _Float16* __restrict__ base, int ld,
                                            int kbase, int lane) {
  int row = lane & 15, hf = lane >> 4;
  const _Float16* p = base + (size_t)row * ld + kbase + hf * 8;
  v8h lo = *(const v8h*)p;
  v8h hi = *(const v8h*)(p + 16);
  v16h r;
#pragma unroll
  for (int i = 0; i < 8; i++) { r[i] = lo[i]; r[i + 8] = hi[i]; }
  return r;
}

// B-frag 32x16 f16: lane l holds column l%16; slot j -> k = 16*(j/8) + 8*(l/16) + (j%8)
__device__ __forceinline__ v16h load_b_frag(const _Float16* __restrict__ rowptr,
                                            int kbase, int lane) {
  const _Float16* p = rowptr + kbase + ((lane >> 4) << 3);
  v8h lo = *(const v8h*)p;
  v8h hi = *(const v8h*)(p + 16);
  v16h r;
#pragma unroll
  for (int i = 0; i < 8; i++) { r[i] = lo[i]; r[i + 8] = hi[i]; }
  return r;
}

__device__ __forceinline__ v8f wmma_f16(v16h a, v16h b, v8f c) {
  return __builtin_amdgcn_wmma_f32_16x16x32_f16(false, a, false, b, (short)0, c, false, false);
}

// ---------- 1: spline -> 4 f16 weight matrices ----------
__global__ void spline4(const float* __restrict__ cq, const float* __restrict__ ck,
                        const float* __restrict__ cv, const float* __restrict__ co,
                        _Float16* __restrict__ wq, _Float16* __restrict__ wk,
                        _Float16* __restrict__ wv, _Float16* __restrict__ wo) {
  int i = blockIdx.x * blockDim.x + threadIdx.x;
  if (i >= WTOT) return;
  double pos = (double)i * ((double)(NCP - 1) / (double)(WTOT - 1));
  int j = (int)pos;
  if (j > NCP - 2) j = NCP - 2;
  float f = (float)(pos - (double)j);
  wq[i] = (_Float16)(cq[j] + f * (cq[j + 1] - cq[j]));
  wk[i] = (_Float16)(ck[j] + f * (ck[j + 1] - ck[j]));
  wv[i] = (_Float16)(cv[j] + f * (cv[j + 1] - cv[j]));
  wo[i] = (_Float16)(co[j] + f * (co[j + 1] - co[j]));
}

// ---------- 2: f32 -> f16 convert ----------
__global__ void cvt_f16(const float* __restrict__ x, _Float16* __restrict__ y, int n4) {
  int i = blockIdx.x * blockDim.x + threadIdx.x;
  if (i >= n4) return;
  float4 v = ((const float4*)x)[i];
  y[i * 4 + 0] = (_Float16)v.x;
  y[i * 4 + 1] = (_Float16)v.y;
  y[i * 4 + 2] = (_Float16)v.z;
  y[i * 4 + 3] = (_Float16)v.w;
}

// ---------- 3: LDS-tiled GEMM: C[M,N] = A[M,K](f16) * W[N,K](f16)^T ----------
// Block 256 thr = 8 waves, tile 128x128, KSTEP=32, double-buffered LDS.
// Wave tile 32x64 (waves arranged 4m x 2n): 8 wmma per K-step.
// mode 0: f16 C[m*N+n]*scale ; mode 1: f32 ; mode 2: f16 V^T per head
#define MBLK 128
#define NBLK 128
#define KSTEP 32

__global__ void __launch_bounds__(256) gemm_lds(const _Float16* __restrict__ A,
                                                const _Float16* __restrict__ W,
                                                void* __restrict__ outp,
                                                int M, int N, int K, float scale, int mode) {
  __shared__ __align__(16) _Float16 Atile[2][MBLK][KSTEP];
  __shared__ __align__(16) _Float16 Btile[2][NBLK][KSTEP];
  int tid = threadIdx.x;
  int lane = tid & 31, wave = tid >> 5;
  int nbx = N / NBLK;
  int mblk = (blockIdx.x / nbx) * MBLK;
  int nblk = (blockIdx.x % nbx) * NBLK;
  int moff = (wave >> 1) * 32, noff = (wave & 1) * 64;
  int nloc = lane & 15, hf = lane >> 4;

  // cooperative staging: 2 threads per row, 16 halves (32B) each
  int lrow = tid >> 1, lcol = (tid & 1) * 16;
  const _Float16* Ag = A + (size_t)(mblk + lrow) * K + lcol;
  const _Float16* Wg = W + (size_t)(nblk + lrow) * K + lcol;

  int nsteps = K / KSTEP;
  v8h ar0, ar1, br0, br1;

  // tile 0 -> LDS buf 0
  ar0 = *(const v8h*)(Ag); ar1 = *(const v8h*)(Ag + 8);
  br0 = *(const v8h*)(Wg); br1 = *(const v8h*)(Wg + 8);
  *(v8h*)&Atile[0][lrow][lcol] = ar0; *(v8h*)&Atile[0][lrow][lcol + 8] = ar1;
  *(v8h*)&Btile[0][lrow][lcol] = br0; *(v8h*)&Btile[0][lrow][lcol + 8] = br1;
  // prefetch tile 1 into registers
  if (nsteps > 1) {
    ar0 = *(const v8h*)(Ag + KSTEP); ar1 = *(const v8h*)(Ag + KSTEP + 8);
    br0 = *(const v8h*)(Wg + KSTEP); br1 = *(const v8h*)(Wg + KSTEP + 8);
  }
  wg_barrier_lds();

  v8f acc[2][4] = {};
  for (int ks = 0; ks < nsteps; ks++) {
    int cur = ks & 1;
    v16h af0 = load_a_frag(&Atile[cur][moff][0], KSTEP, 0, lane);
    v16h af1 = load_a_frag(&Atile[cur][moff + 16][0], KSTEP, 0, lane);
#pragma unroll
    for (int t = 0; t < 4; t++) {
      v16h bf = load_b_frag(&Btile[cur][noff + t * 16 + nloc][0], 0, lane);
      acc[0][t] = wmma_f16(af0, bf, acc[0][t]);
      acc[1][t] = wmma_f16(af1, bf, acc[1][t]);
    }
    if (ks + 1 < nsteps) {
      int nxt = cur ^ 1;
      *(v8h*)&Atile[nxt][lrow][lcol] = ar0; *(v8h*)&Atile[nxt][lrow][lcol + 8] = ar1;
      *(v8h*)&Btile[nxt][lrow][lcol] = br0; *(v8h*)&Btile[nxt][lrow][lcol + 8] = br1;
      if (ks + 2 < nsteps) {
        int kb = (ks + 2) * KSTEP;
        ar0 = *(const v8h*)(Ag + kb); ar1 = *(const v8h*)(Ag + kb + 8);
        br0 = *(const v8h*)(Wg + kb); br1 = *(const v8h*)(Wg + kb + 8);
      }
      wg_barrier_lds();   // uniform: executed by all waves whenever more steps remain
    }
  }

  // store
#pragma unroll
  for (int mf = 0; mf < 2; mf++)
#pragma unroll
    for (int t = 0; t < 4; t++)
#pragma unroll
      for (int v = 0; v < 8; v++) {
        int m = mblk + moff + mf * 16 + v + 8 * hf;
        int n = nblk + noff + t * 16 + nloc;
        float val = acc[mf][t][v] * scale;
        if (mode == 0) {
          ((_Float16*)outp)[(size_t)m * N + n] = (_Float16)val;
        } else if (mode == 1) {
          ((float*)outp)[(size_t)m * N + n] = val;
        } else {
          int b = m >> 11, s = m & 2047;   // SEQ = 2048
          int h = n >> 7, d = n & 127;     // HDIM = 128
          ((_Float16*)outp)[(((size_t)(b * NHEADS + h) * HDIM + d) << 11) + s] = (_Float16)val;
        }
      }
}

// ---------- 4: causal flash attention with LDS-staged K/V tiles ----------
// Block 128 thr = 4 waves, each wave owns 16 query rows (64 per block).
// All waves run the same number of 32-key steps (fully-masked tiles are benign:
// alpha=1, P=0), enabling cooperative double-buffered K/V staging in LDS.
__global__ void __launch_bounds__(128) flash_attn(const _Float16* __restrict__ Q,
                                                  const _Float16* __restrict__ Kh,
                                                  const _Float16* __restrict__ VT,
                                                  _Float16* __restrict__ O) {
  __shared__ __align__(16) _Float16 Ktile[2][32][HDIM];   // [key][d]
  __shared__ __align__(16) _Float16 Vtile[2][HDIM][32];   // [d][key]  (V^T slice)
  __shared__ __align__(16) _Float16 plds[4][16][32];      // per-wave P relayout

  int tid = threadIdx.x;
  int lane = tid & 31, wave = tid >> 5;
  int nqb = SEQ / 64;
  int bh = blockIdx.x / nqb, qblk = blockIdx.x % nqb;
  int b = bh / NHEADS, h = bh % NHEADS;
  int qbase = qblk * 64 + wave * 16;
  int hf = lane >> 4, nloc = lane & 15;

  const _Float16* Qt = Q + ((size_t)(b * SEQ + qbase)) * HIDDEN + h * HDIM;
  const _Float16* Kt = Kh + ((size_t)b * SEQ) * HIDDEN + h * HDIM;
  const _Float16* Vt = VT + ((size_t)bh * HDIM) * SEQ;

  // cooperative staging indices
  int krow = tid >> 2, kcol = (tid & 3) * 32;   // K tile: 32 rows x 128 halves
  const _Float16* Kg = Kt + (size_t)krow * HIDDEN + kcol;
  const _Float16* Vg = Vt + (size_t)tid * SEQ;  // V tile: 128 rows x 32 halves

  v16h qf[4];
#pragma unroll
  for (int dc = 0; dc < 4; dc++) qf[dc] = load_a_frag(Qt, HIDDEN, dc * 32, lane);

  v8f o[8] = {};
  float mrow[8], lrow_[8];
#pragma unroll
  for (int v = 0; v < 8; v++) { mrow[v] = -1e30f; lrow_[v] = 0.f; }

  int nsteps = 2 * qblk + 2;   // keys 0 .. qblk*64+63, uniform across waves

  v8h kr[4], vr[4];
  // tile 0 -> LDS buf 0
#pragma unroll
  for (int i = 0; i < 4; i++) { kr[i] = *(const v8h*)(Kg + i * 8); vr[i] = *(const v8h*)(Vg + i * 8); }
#pragma unroll
  for (int i = 0; i < 4; i++) {
    *(v8h*)&Ktile[0][krow][kcol + i * 8] = kr[i];
    *(v8h*)&Vtile[0][tid][i * 8] = vr[i];
  }
  if (nsteps > 1) {
#pragma unroll
    for (int i = 0; i < 4; i++) {
      kr[i] = *(const v8h*)(Kg + 32 * HIDDEN + i * 8);
      vr[i] = *(const v8h*)(Vg + 32 + i * 8);
    }
  }
  wg_barrier_lds();

  for (int kt = 0; kt < nsteps; kt++) {
    int cur = kt & 1;
    int kk0 = kt << 5;
    v8f sc[2] = {};
#pragma unroll
    for (int nt = 0; nt < 2; nt++)
#pragma unroll
      for (int dc = 0; dc < 4; dc++) {
        v16h bf = load_b_frag(&Ktile[cur][nt * 16 + nloc][0], dc * 32, lane);
        sc[nt] = wmma_f16(qf[dc], bf, sc[nt]);
      }
    if (kk0 + 31 >= qbase) {  // diagonal or fully-masked tiles
#pragma unroll
      for (int nt = 0; nt < 2; nt++)
#pragma unroll
        for (int v = 0; v < 8; v++) {
          int kk = kk0 + nt * 16 + nloc;
          int q = qbase + v + 8 * hf;
          if (kk > q) sc[nt][v] = -1e30f;
        }
    }
    float alpha[8];
#pragma unroll
    for (int v = 0; v < 8; v++) {
      float t = fmaxf(sc[0][v], sc[1][v]);
#pragma unroll
      for (int msk = 1; msk < 16; msk <<= 1) t = fmaxf(t, __shfl_xor(t, msk, 32));
      float nm = fmaxf(mrow[v], t);
      alpha[v] = __expf(mrow[v] - nm);
      mrow[v] = nm;
      sc[0][v] = __expf(sc[0][v] - nm);
      sc[1][v] = __expf(sc[1][v] - nm);
      float rs = sc[0][v] + sc[1][v];
#pragma unroll
      for (int msk = 1; msk < 16; msk <<= 1) rs += __shfl_xor(rs, msk, 32);
      lrow_[v] = lrow_[v] * alpha[v] + rs;
    }
#pragma unroll
    for (int dt = 0; dt < 8; dt++)
#pragma unroll
      for (int v = 0; v < 8; v++) o[dt][v] *= alpha[v];

    // D-layout -> A-layout relayout of P through per-wave LDS (same-wave DS in-order)
#pragma unroll
    for (int v = 0; v < 8; v++) {
      plds[wave][v + 8 * hf][nloc]      = (_Float16)sc[0][v];
      plds[wave][v + 8 * hf][16 + nloc] = (_Float16)sc[1][v];
    }
    v16h pf;
    {
      const _Float16* p = &plds[wave][nloc][hf * 8];
      v8h lo = *(const v8h*)p;
      v8h hi = *(const v8h*)(p + 16);
#pragma unroll
      for (int i = 0; i < 8; i++) { pf[i] = lo[i]; pf[i + 8] = hi[i]; }
    }
#pragma unroll
    for (int dt = 0; dt < 8; dt++) {
      v16h vf = load_b_frag(&Vtile[cur][dt * 16 + nloc][0], 0, lane);
      o[dt] = wmma_f16(pf, vf, o[dt]);
    }

    if (kt + 1 < nsteps) {
      int nxt = cur ^ 1;
#pragma unroll
      for (int i = 0; i < 4; i++) {
        *(v8h*)&Ktile[nxt][krow][kcol + i * 8] = kr[i];
        *(v8h*)&Vtile[nxt][tid][i * 8] = vr[i];
      }
      if (kt + 2 < nsteps) {
        int kk2 = (kt + 2) * 32;
#pragma unroll
        for (int i = 0; i < 4; i++) {
          kr[i] = *(const v8h*)(Kg + (size_t)kk2 * HIDDEN + i * 8);
          vr[i] = *(const v8h*)(Vg + kk2 + i * 8);
        }
      }
      wg_barrier_lds();   // uniform across the block
    }
  }

  float rinv[8];
#pragma unroll
  for (int v = 0; v < 8; v++) rinv[v] = 1.0f / lrow_[v];
  _Float16* Ot = O + ((size_t)(b * SEQ + qbase)) * HIDDEN + h * HDIM;
#pragma unroll
  for (int dt = 0; dt < 8; dt++)
#pragma unroll
    for (int v = 0; v < 8; v++) {
      int m = v + 8 * hf;
      int d = dt * 16 + nloc;
      Ot[(size_t)m * HIDDEN + d] = (_Float16)(o[dt][v] * rinv[v]);
    }
}

// ---------- launch ----------
extern "C" void kernel_launch(void* const* d_in, const int* in_sizes, int n_in,
                              void* d_out, int out_size, void* d_ws, size_t ws_size,
                              hipStream_t stream) {
  (void)in_sizes; (void)n_in; (void)out_size; (void)ws_size;
  const float* x  = (const float*)d_in[0];
  const float* cq = (const float*)d_in[1];
  const float* ck = (const float*)d_in[2];
  const float* cv = (const float*)d_in[3];
  const float* co = (const float*)d_in[4];
  // d_in[5] (attention_mask) is deterministic causal; applied analytically.

  char* ws = (char*)d_ws;
  size_t off = 0;
  auto alloc = [&](size_t bytes) { char* p = ws + off; off += (bytes + 255) & ~(size_t)255; return p; };
  _Float16* Wq = (_Float16*)alloc((size_t)WTOT * 2);
  _Float16* Wk = (_Float16*)alloc((size_t)WTOT * 2);
  _Float16* Wv = (_Float16*)alloc((size_t)WTOT * 2);
  _Float16* Wo = (_Float16*)alloc((size_t)WTOT * 2);
  _Float16* Xh = (_Float16*)alloc((size_t)MTOT * HIDDEN * 2);
  _Float16* Qh = (_Float16*)alloc((size_t)MTOT * HIDDEN * 2);
  _Float16* Kh = (_Float16*)alloc((size_t)MTOT * HIDDEN * 2);
  _Float16* VT = (_Float16*)alloc((size_t)MTOT * HIDDEN * 2);
  _Float16* At = (_Float16*)alloc((size_t)MTOT * HIDDEN * 2);

  spline4<<<(WTOT + 255) / 256, 256, 0, stream>>>(cq, ck, cv, co, Wq, Wk, Wv, Wo);

  int n4 = (MTOT * HIDDEN) / 4;
  cvt_f16<<<(n4 + 255) / 256, 256, 0, stream>>>(x, Xh, n4);

  int gemmBlocks = (MTOT / MBLK) * (HIDDEN / NBLK);   // 32*16 = 512
  const float qscale = 0.08838834764831845f;           // 1/sqrt(128)
  gemm_lds<<<gemmBlocks, 256, 0, stream>>>(Xh, Wq, Qh, MTOT, HIDDEN, HIDDEN, qscale, 0);
  gemm_lds<<<gemmBlocks, 256, 0, stream>>>(Xh, Wk, Kh, MTOT, HIDDEN, HIDDEN, 1.0f, 0);
  gemm_lds<<<gemmBlocks, 256, 0, stream>>>(Xh, Wv, VT, MTOT, HIDDEN, HIDDEN, 1.0f, 2);

  flash_attn<<<BATCH * NHEADS * (SEQ / 64), 128, 0, stream>>>(Qh, Kh, VT, At);

  gemm_lds<<<gemmBlocks, 256, 0, stream>>>(At, Wo, d_out, MTOT, HIDDEN, HIDDEN, 1.0f, 1);
}
